// Net_66537633350281
// MI455X (gfx1250) — matrix operand order repaired and checked by reading, not compiled, not run on hardware
//
#include <hip/hip_runtime.h>
#include <hip/hip_bf16.h>
#include <stdint.h>

#define NN 100000
#define NG 100

typedef __bf16 bf16_t;
typedef __attribute__((ext_vector_type(16))) __bf16 v16bf;
typedef __attribute__((ext_vector_type(8)))  __bf16 v8bf;
typedef __attribute__((ext_vector_type(8)))  float  v8f;

__device__ __forceinline__ void atomicAddF(float* p, float v) {
  __hip_atomic_fetch_add(p, v, __ATOMIC_RELAXED, __HIP_MEMORY_SCOPE_AGENT);
}
// sign-aware integer atomics implement float max/min with hardware atomics
__device__ __forceinline__ void atomicMaxF(float* a, float v) {
  if (v >= 0.f) atomicMax((int*)a, __float_as_int(v));
  else          atomicMin((unsigned int*)a, __float_as_uint(v));
}
__device__ __forceinline__ void atomicMinF(float* a, float v) {
  if (v < 0.f) atomicMax((unsigned int*)a, __float_as_uint(v));
  else         atomicMin((int*)a, __float_as_int(v));
}

__global__ void k_zero_i32(int* p, int n) {
  int t = blockIdx.x * blockDim.x + threadIdx.x;
  if (t < n) p[t] = 0;
}
__global__ void k_zero_f32(float* p, int n) {
  int t = blockIdx.x * blockDim.x + threadIdx.x;
  if (t < n) p[t] = 0.f;
}
__global__ void k_edge_deg(const int* __restrict__ dst, int* __restrict__ deg, int E) {
  int t = blockIdx.x * blockDim.x + threadIdx.x;
  if (t < E) atomicAdd(&deg[dst[t]], 1);
}
__global__ void k_dinv(const int* __restrict__ deg, float* __restrict__ dinv, int n) {
  int t = blockIdx.x * blockDim.x + threadIdx.x;
  if (t < n) dinv[t] = rsqrtf((float)(deg[t] + 1));  // +1 self loop
}

// conv1 input GEMM: [NN,5] @ [5,32] -> xw (K=5 too small for WMMA; VALU is free here)
__global__ void k_xw1(const float* __restrict__ x, const float* __restrict__ W1,
                      float* __restrict__ xw) {
  int t = blockIdx.x * blockDim.x + threadIdx.x;
  if (t >= NN * 32) return;
  int n = t >> 5, o = t & 31;
  const float* xr = x + n * 5;
  float acc = 0.f;
#pragma unroll
  for (int k = 0; k < 5; ++k) acc = fmaf(xr[k], W1[k * 32 + o], acc);
  xw[t] = acc;
}

// edge aggregation: agg[dst] += xw[src] * dinv[src]*dinv[dst]  (L2-resident atomics)
template<int F>
__global__ void k_scatter(const int* __restrict__ src, const int* __restrict__ dst,
                          const float* __restrict__ dinv, const float* __restrict__ xw,
                          float* __restrict__ agg, int E) {
  int e = blockIdx.x * blockDim.x + threadIdx.x;
  if (e >= E) return;
  int s = src[e], d = dst[e];
  float nrm = dinv[s] * dinv[d];
  const float4* xs = (const float4*)(xw + (size_t)s * F);
  float* ag = agg + (size_t)d * F;
#pragma unroll
  for (int f = 0; f < F / 4; ++f) {
    float4 v = xs[f];
    atomicAddF(ag + 4 * f + 0, v.x * nrm);
    atomicAddF(ag + 4 * f + 1, v.y * nrm);
    atomicAddF(ag + 4 * f + 2, v.z * nrm);
    atomicAddF(ag + 4 * f + 3, v.w * nrm);
  }
}

// add self-loop term + bias, relu, convert to bf16 for next WMMA GEMM
template<int F>
__global__ void k_finish(const float* __restrict__ agg, const float* __restrict__ xw,
                         const float* __restrict__ dinv, const float* __restrict__ bias,
                         bf16_t* __restrict__ out) {
  int t = blockIdx.x * blockDim.x + threadIdx.x;
  if (t >= NN * F) return;
  int n = t / F, f = t % F;
  float dv = dinv[n];
  float v = agg[t] + xw[t] * dv * dv + bias[f];
  out[t] = (bf16_t)fmaxf(v, 0.f);
}

__global__ void k_wbf16(const float* __restrict__ W2, const float* __restrict__ nn1W,
                        bf16_t* __restrict__ W2bf, bf16_t* __restrict__ nn1bf) {
  int t = blockIdx.x * blockDim.x + threadIdx.x;
  if (t < 32 * 64) W2bf[t] = (bf16_t)W2[t];
  if (t < 64 * 128) {  // pad 124 -> 128 with zeros
    int r = t >> 7, c = t & 127;
    nn1bf[t] = (c < 124) ? (bf16_t)nn1W[r * 124 + c] : (bf16_t)0.f;
  }
}

// one wave computes one 16x16 f32 tile of A[M,K](bf16) @ B[K,NB](bf16)
template<int K, int NB>
__device__ __forceinline__ v8f wmma_tile(const bf16_t* __restrict__ A,
                                         const bf16_t* __restrict__ B,
                                         int tM, int tN, int lane) {
  int l = lane & 15, hi = lane >> 4;
  int row = tM * 16 + l;
  int col = tN * 16 + l;
  v8f c = {};
#pragma unroll
  for (int kk = 0; kk < K; kk += 32) {
    // A 16x32 bf16: lanes 0-15 K{0..7,16..23}, lanes 16-31 K{8..15,24..31}
    v8bf alo = *(const v8bf*)(A + (size_t)row * K + kk + hi * 8);
    v8bf ahi = *(const v8bf*)(A + (size_t)row * K + kk + 16 + hi * 8);
    v16bf a;
#pragma unroll
    for (int i = 0; i < 8; ++i) { a[i] = alo[i]; a[8 + i] = ahi[i]; }
    // B 32x16 bf16: lanes 0-15 K{0..7,16..23}, lanes 16-31 K{8..15,24..31}, col = lane&15
    v16bf b;
#pragma unroll
    for (int j = 0; j < 16; ++j) b[j] = B[(size_t)(kk + hi * 8 + (j & 7) + (j >> 3) * 16) * NB + col];
    c = __builtin_amdgcn_wmma_f32_16x16x32_bf16(false, a, false, b, (short)0, c,
                                                false, false);
  }
  return c;
}

// conv2 GEMM: C[NN,64] = h1[NN,32] @ W2[32,64]
__global__ void k_gemm_conv2(const bf16_t* __restrict__ A, const bf16_t* __restrict__ B,
                             float* __restrict__ C) {
  int lane = threadIdx.x & 31;
  int tile = blockIdx.x * 4 + (threadIdx.x >> 5);
  int tN = tile & 3, tM = tile >> 2;
  v8f c = wmma_tile<32, 64>(A, B, tM, tN, lane);
  int l = lane & 15, hi = lane >> 4;
  int col = tN * 16 + l;
#pragma unroll
  for (int r = 0; r < 8; ++r)
    C[(size_t)(tM * 16 + hi * 8 + r) * 64 + col] = c[r];
}

// fused: h3 = h2[NN,64] @ nn1_W[64,124] + b, pooled (max/min/sum) per graph.
// Never materializes h3 (saves ~100MB of HBM round-trip).
__global__ void k_gemm_pool(const bf16_t* __restrict__ A, const bf16_t* __restrict__ B,
                            const float* __restrict__ bias, const int* __restrict__ batch,
                            float* __restrict__ pmax, float* __restrict__ pmin,
                            float* __restrict__ psum) {
  int lane = threadIdx.x & 31;
  int tile = blockIdx.x * 4 + (threadIdx.x >> 5);
  int tN = tile & 7, tM = tile >> 3;
  v8f c = wmma_tile<64, 128>(A, B, tM, tN, lane);
  int l = lane & 15, hi = lane >> 4;
  int col = tN * 16 + l;
  if (col < 124) {
    float bb = bias[col];
#pragma unroll
    for (int r = 0; r < 8; ++r) {
      int row = tM * 16 + hi * 8 + r;
      float v = c[r] + bb;
      int off = batch[row] * 124 + col;
      atomicAddF(psum + off, v);
      atomicMaxF(pmax + off, v);
      atomicMinF(pmin + off, v);
    }
  }
}

__global__ void k_pool_init(float* pmax, float* pmin, float* psum, int* cnt) {
  int t = blockIdx.x * blockDim.x + threadIdx.x;
  if (t < NG * 124) { pmax[t] = -3.402823e38f; pmin[t] = 3.402823e38f; psum[t] = 0.f; }
  if (t < NG) cnt[t] = 0;
}
__global__ void k_count(const int* __restrict__ batch, int* __restrict__ cnt) {
  int t = blockIdx.x * blockDim.x + threadIdx.x;
  if (t < NN) atomicAdd(&cnt[batch[t]], 1);
}

// whole MLP head on the 100 pooled rows in one block (conv4 with 2 self loops
// per node reduces to g@W4 + b4 exactly: deg=2, norm=1/2 twice -> identity).
__global__ void k_final(const float* __restrict__ pmax, const float* __restrict__ pmin,
                        const float* __restrict__ psum, const int* __restrict__ cnt,
                        const float* __restrict__ nn2W, const float* __restrict__ nn2b,
                        const float* __restrict__ W4, const float* __restrict__ b4,
                        const float* __restrict__ nn3W, const float* __restrict__ nn3b,
                        const float* __restrict__ nn4W, const float* __restrict__ nn4b,
                        float* __restrict__ out) {
  __shared__ float g1[NG * 64];
  __shared__ float g2[NG * 32];
  __shared__ float g3[NG * 16];
  int t = threadIdx.x;
  for (int idx = t; idx < NG * 64; idx += blockDim.x) {
    int g = idx >> 6, o = idx & 63;
    float cn = (float)cnt[g];
    bool ne = cn > 0.f;
    float inv = 1.f / fmaxf(cn, 1.f);
    float acc = nn2b[o];
    for (int f = 0; f < 124; ++f) {
      int off = g * 124 + f;
      float mx = ne ? pmax[off] : 0.f;
      float mn = ne ? pmin[off] : 0.f;
      float sm = psum[off];
      float mean = sm * inv;
      acc = fmaf(fmaxf(mx, 0.f),   nn2W[f * 64 + o],         acc);
      acc = fmaf(fmaxf(mn, 0.f),   nn2W[(124 + f) * 64 + o], acc);
      acc = fmaf(fmaxf(sm, 0.f),   nn2W[(248 + f) * 64 + o], acc);
      acc = fmaf(fmaxf(mean, 0.f), nn2W[(372 + f) * 64 + o], acc);
    }
    g1[idx] = acc;
  }
  __syncthreads();
  for (int idx = t; idx < NG * 32; idx += blockDim.x) {
    int g = idx >> 5, o = idx & 31;
    float acc = b4[o];
    for (int j = 0; j < 64; ++j) acc = fmaf(g1[g * 64 + j], W4[j * 32 + o], acc);
    g2[idx] = fmaxf(acc, 0.f);  // relu(conv4)
  }
  __syncthreads();
  for (int idx = t; idx < NG * 16; idx += blockDim.x) {
    int g = idx >> 4, o = idx & 15;
    float acc = nn3b[o];
    for (int j = 0; j < 32; ++j) acc = fmaf(g2[g * 32 + j], nn3W[j * 16 + o], acc);
    g3[idx] = fmaxf(acc, 0.f);
  }
  __syncthreads();
  for (int g = t; g < NG; g += blockDim.x) {
    float acc = nn4b[0];
    for (int j = 0; j < 16; ++j) acc = fmaf(g3[g * 16 + j], nn4W[j], acc);
    out[g] = acc;
  }
}

extern "C" void kernel_launch(void* const* d_in, const int* in_sizes, int n_in,
                              void* d_out, int out_size, void* d_ws, size_t ws_size,
                              hipStream_t stream) {
  const float* x    = (const float*)d_in[0];
  const int*   edge = (const int*)d_in[1];
  const int*   batch= (const int*)d_in[2];
  const float* W1   = (const float*)d_in[3];
  const float* b1   = (const float*)d_in[4];
  const float* W2   = (const float*)d_in[5];
  const float* b2   = (const float*)d_in[6];
  const float* nn1W = (const float*)d_in[7];
  const float* nn1b = (const float*)d_in[8];
  const float* nn2W = (const float*)d_in[9];
  const float* nn2b = (const float*)d_in[10];
  const float* W4   = (const float*)d_in[11];
  const float* b4   = (const float*)d_in[12];
  const float* nn3W = (const float*)d_in[13];
  const float* nn3b = (const float*)d_in[14];
  const float* nn4W = (const float*)d_in[15];
  const float* nn4b = (const float*)d_in[16];
  float* out = (float*)d_out;
  (void)n_in; (void)out_size; (void)ws_size;

  const int E = in_sizes[1] / 2;
  const int* src = edge;
  const int* dst = edge + E;

  char* ws = (char*)d_ws;
  size_t off = 0;
  auto alloc = [&](size_t bytes) {
    char* p = ws + off;
    off = (off + bytes + 255) & ~(size_t)255;
    return p;
  };
  float*  dinv  = (float*)alloc((size_t)NN * 4);
  int*    deg   = (int*)alloc((size_t)NN * 4);
  float*  xw    = (float*)alloc((size_t)NN * 64 * 4);   // conv1 (32) / conv2 (64)
  float*  agg   = (float*)alloc((size_t)NN * 64 * 4);
  bf16_t* hbf   = (bf16_t*)alloc((size_t)NN * 64 * 2);  // h1 (32) then h2 (64)
  bf16_t* W2bf  = (bf16_t*)alloc(32 * 64 * 2);
  bf16_t* nn1bf = (bf16_t*)alloc(64 * 128 * 2);
  float*  pmax  = (float*)alloc(NG * 124 * 4);
  float*  pmin  = (float*)alloc(NG * 124 * 4);
  float*  psum  = (float*)alloc(NG * 124 * 4);
  int*    cnt   = (int*)alloc(NG * 4);

  const int B = 256;
  auto gs = [](int n, int b) { return (n + b - 1) / b; };

  // degrees / normalization
  k_zero_i32<<<gs(NN, B), B, 0, stream>>>(deg, NN);
  k_edge_deg<<<gs(E, B), B, 0, stream>>>(dst, deg, E);
  k_dinv<<<gs(NN, B), B, 0, stream>>>(deg, dinv, NN);

  // conv1
  k_xw1<<<gs(NN * 32, B), B, 0, stream>>>(x, W1, xw);
  k_zero_f32<<<gs(NN * 32, B), B, 0, stream>>>(agg, NN * 32);
  k_scatter<32><<<gs(E, B), B, 0, stream>>>(src, dst, dinv, xw, agg, E);
  k_finish<32><<<gs(NN * 32, B), B, 0, stream>>>(agg, xw, dinv, b1, hbf);

  // bf16 weights for WMMA
  k_wbf16<<<gs(64 * 128, B), B, 0, stream>>>(W2, nn1W, W2bf, nn1bf);

  // conv2: WMMA GEMM then scatter
  k_gemm_conv2<<<(NN / 16) * 4 / 4, 128, 0, stream>>>(hbf, W2bf, xw);
  k_zero_f32<<<gs(NN * 64, B), B, 0, stream>>>(agg, NN * 64);
  k_scatter<64><<<gs(E, B), B, 0, stream>>>(src, dst, dinv, xw, agg, E);
  k_finish<64><<<gs(NN * 64, B), B, 0, stream>>>(agg, xw, dinv, b2, hbf);

  // pooling: fused WMMA GEMM (h2 @ nn1_W + b) -> segment max/min/sum
  k_pool_init<<<gs(NG * 124, B), B, 0, stream>>>(pmax, pmin, psum, cnt);
  k_count<<<gs(NN, B), B, 0, stream>>>(batch, cnt);
  k_gemm_pool<<<(NN / 16) * 8 / 4, 128, 0, stream>>>(hbf, nn1bf, nn1b, batch,
                                                     pmax, pmin, psum);

  // MLP head on 100 graphs
  k_final<<<1, 256, 0, stream>>>(pmax, pmin, psum, cnt, nn2W, nn2b, W4, b4,
                                 nn3W, nn3b, nn4W, nn4b, out);
}
